// NoSemanticSchemaGNN_15522011807981
// MI455X (gfx1250) — matrix operand order repaired and checked by reading, not compiled, not run on hardware
//
#include <hip/hip_runtime.h>
#include <hip/hip_bf16.h>
#include <math.h>

#define NN   30000
#define EE   40000
#define RR   5
#define BB   64
#define IND  20
#define HIDC 256
#define HH   4
#define LL   3
#define DD   64

typedef __attribute__((ext_vector_type(16))) __bf16 v16bf;
typedef __attribute__((ext_vector_type(8)))  float  v8f;

union FragU {
  v16bf v;
  unsigned short h[16];
  uint4 q[2];
};

__device__ __forceinline__ unsigned short f2bf(float f) {
  unsigned int u = __float_as_uint(f);
  unsigned int r = u + 0x7fffu + ((u >> 16) & 1u);   // round-to-nearest-even
  return (unsigned short)(r >> 16);
}
__device__ __forceinline__ float geluf(float x) {
  return 0.5f * x * (1.0f + erff(x * 0.70710678118654752f));
}
__device__ __forceinline__ float wave_sum32(float v) {
  #pragma unroll
  for (int o = 16; o >= 1; o >>= 1) v += __shfl_xor(v, o);
  return v;
}
__device__ __forceinline__ void atomicMaxF(float* addr, float val) {
  // IEEE ordering trick: works for mixed signs with -inf init
  if (val >= 0.0f) atomicMax((int*)addr, __float_as_int(val));
  else             atomicMin((unsigned int*)addr, __float_as_uint(val));
}

// ---------------------------------------------------------------------------
// Generic bf16-WMMA GEMM: out[M,Nn] = OUT_ACT( IN_ACT(A[M,K]) @ Wb[K,Nn] + bias )
// Block = 128 threads (4 waves), 64x64 output tile, K stepped by 32.
// IN_ACT: 0=none 1=gelu (applied to A elements); OUT_ACT: 0=none 1=relu.
// Nn must be a multiple of 64 (true for all call sites: 256/128).
// Fast path: fully in-bounds tiles use b128 global loads; scalar path only for
// fringe tiles (last row block / K not multiple of 32).
// ---------------------------------------------------------------------------
template <int IN_ACT, int OUT_ACT>
__global__ void gemm_bf16_wmma(const float* __restrict__ A,
                               const unsigned short* __restrict__ Wb,
                               const float* __restrict__ bias,
                               float* __restrict__ out,
                               int M, int K, int Nn)
{
  __shared__ unsigned short Ash[64 * 32];   // [row][k]   (bf16)
  __shared__ unsigned short Bsh[64 * 32];   // [col][k]   (bf16, transposed)

  const int t    = threadIdx.x;     // 0..127
  const int wv   = t >> 5;          // wave 0..3 -> rows [16*wv, 16*wv+16)
  const int lane = t & 31;
  const int hlf  = lane >> 4;       // 0/1
  const int l15  = lane & 15;

  const int rowBase = blockIdx.y * 64;
  const int colBase = blockIdx.x * 64;

  v8f acc[4];
  #pragma unroll
  for (int n = 0; n < 4; ++n) acc[n] = v8f{0.f,0.f,0.f,0.f,0.f,0.f,0.f,0.f};

  const int arow  = t >> 1;         // 0..63
  const int akoff = (t & 1) * 16;   // 0/16
  const int bkrow = t >> 2;         // 0..31
  const int bcoff = (t & 3) * 16;   // 0/16/32/48

  for (int k0 = 0; k0 < K; k0 += 32) {
    // gfx1250 global_prefetch_b8 for next A tile (near caches: consumed soon)
    if (k0 + 32 < K) {
      int gr = rowBase + arow;
      if (gr < M) __builtin_prefetch(&A[(size_t)gr * K + k0 + 32 + akoff], 0, 3);
    }
    __syncthreads();
    { // A tile: fp32 -> bf16 (+ optional compile-time gelu)
      int gr = rowBase + arow;
      unsigned short* dstp = &Ash[arow * 32 + akoff];
      if (gr < M && k0 + 32 <= K) {
        // fast path: 4x global_load_b128, pack pairs, 8B LDS stores
        const float* src = A + (size_t)gr * K + k0 + akoff;
        #pragma unroll
        for (int j4 = 0; j4 < 4; ++j4) {
          float4 f = *(const float4*)(src + j4 * 4);
          if (IN_ACT == 1) {
            f.x = geluf(f.x); f.y = geluf(f.y);
            f.z = geluf(f.z); f.w = geluf(f.w);
          }
          uint2 p;
          p.x = (unsigned)f2bf(f.x) | ((unsigned)f2bf(f.y) << 16);
          p.y = (unsigned)f2bf(f.z) | ((unsigned)f2bf(f.w) << 16);
          *(uint2*)(dstp + j4 * 4) = p;
        }
      } else {
        #pragma unroll
        for (int j = 0; j < 16; ++j) {
          int kk = k0 + akoff + j;
          float v = (gr < M && kk < K) ? A[(size_t)gr * K + kk] : 0.0f;
          if (IN_ACT == 1) v = geluf(v);
          dstp[j] = f2bf(v);
        }
      }
    }
    { // B tile transposed into LDS: Bsh[col][k]
      int gk = k0 + bkrow;
      union { uint4 q[2]; unsigned short u[16]; } wreg;
      if (gk < K) {
        const unsigned short* src = Wb + (size_t)gk * Nn + colBase + bcoff;
        wreg.q[0] = *(const uint4*)(src);
        wreg.q[1] = *(const uint4*)(src + 8);
      } else {
        wreg.q[0] = uint4{0u, 0u, 0u, 0u};
        wreg.q[1] = uint4{0u, 0u, 0u, 0u};
      }
      #pragma unroll
      for (int j = 0; j < 16; ++j)
        Bsh[(bcoff + j) * 32 + bkrow] = wreg.u[j];
    }
    __syncthreads();

    // A fragment: lanes 0-15 -> K 0-15, lanes 16-31 -> K 16-31 (contiguous, matches B)
    FragU fa;
    const unsigned short* ap = &Ash[(wv * 16 + l15) * 32 + hlf * 16];
    fa.q[0] = *(const uint4*)(ap);
    fa.q[1] = *(const uint4*)(ap + 8);

    #pragma unroll
    for (int n = 0; n < 4; ++n) {
      // B fragment: col = l15 + 16n, 16 contiguous K at hlf*16
      FragU fb;
      const uint4* bp = (const uint4*)&Bsh[(n * 16 + l15) * 32 + hlf * 16];
      fb.q[0] = bp[0];
      fb.q[1] = bp[1];
      acc[n] = __builtin_amdgcn_wmma_f32_16x16x32_bf16(
          false, fa.v, false, fb.v, (short)0, acc[n], false, false);
    }
  }

  // D layout: M = v + 8*hlf (within wave tile), N = l15
  const int rb = rowBase + wv * 16 + hlf * 8;
  #pragma unroll
  for (int n = 0; n < 4; ++n) {
    int c = colBase + n * 16 + l15;
    float bval = bias[c];
    #pragma unroll
    for (int v = 0; v < 8; ++v) {
      int r = rb + v;
      if (r < M) {
        float val = acc[n][v] + bval;
        if (OUT_ACT == 1) val = fmaxf(val, 0.0f);
        out[(size_t)r * Nn + c] = val;
      }
    }
  }
}

// ---------------------------------------------------------------------------
// Edge pass A: alpha[r,e,h] = (k[src] @ a_rel[r,h]) . q[dst] * p_rel/sqrt(D)
// and segment-max into m[dst,h]. blockIdx.y = r*H+h, a_rel tile in LDS.
// ---------------------------------------------------------------------------
__global__ void edge_alpha_kernel(const float* __restrict__ kbuf,
                                  const float* __restrict__ qbuf,
                                  const int* __restrict__ ei,
                                  const float* __restrict__ arel_l,
                                  const float* __restrict__ prel_l,
                                  float* __restrict__ alpha,
                                  float* __restrict__ m)
{
  __shared__ float Ar[DD * DD];
  const int rh = blockIdx.y;
  const int r  = rh >> 2;     // H == 4
  const int hh = rh & 3;
  const float* Ag = arel_l + (size_t)rh * DD * DD;
  for (int i = threadIdx.x; i < DD * DD; i += blockDim.x) Ar[i] = Ag[i];
  __syncthreads();

  int e = blockIdx.x * blockDim.x + threadIdx.x;
  if (e >= EE) return;
  int sn = ei[(r * 2)     * EE + e];
  int dn = ei[(r * 2 + 1) * EE + e];

  float kv[DD];
  const float* kr = kbuf + (size_t)sn * HIDC + hh * DD;
  #pragma unroll
  for (int d = 0; d < DD; ++d) kv[d] = kr[d];
  const float* qr = qbuf + (size_t)dn * HIDC + hh * DD;

  float acc = 0.0f;
  for (int f = 0; f < DD; ++f) {
    float kp = 0.0f;
    #pragma unroll
    for (int d = 0; d < DD; ++d) kp += kv[d] * Ar[d * DD + f];
    acc += kp * qr[f];
  }
  acc *= prel_l[rh] * 0.125f;   // 1/sqrt(64)
  alpha[((size_t)r * EE + e) * HH + hh] = acc;
  atomicMaxF(&m[dn * HH + hh], acc);
}

__global__ void fix_m_kernel(float* m, int n) {
  int i = blockIdx.x * blockDim.x + threadIdx.x;
  if (i < n) { float v = m[i]; if (isinf(v)) m[i] = 0.0f; }
}

// Edge pass B: e = exp(alpha - m[dst]); segment-sum into s[dst,h]
__global__ void edge_exp_kernel(const int* __restrict__ ei,
                                const float* __restrict__ m,
                                float* __restrict__ alpha,
                                float* __restrict__ s)
{
  int idx = blockIdx.x * blockDim.x + threadIdx.x;
  if (idx >= RR * EE * HH) return;
  int r   = idx / (EE * HH);
  int rem = idx - r * (EE * HH);
  int e   = rem / HH;
  int hh  = rem - e * HH;
  int dn  = ei[(r * 2 + 1) * EE + e];
  float ev = expf(alpha[idx] - m[dn * HH + hh]);
  alpha[idx] = ev;
  atomicAdd(&s[dn * HH + hh], ev);
}

// Edge pass C: agg[dst] += (e/s[dst]) * (v[src] @ m_rel[r,h])
__global__ void edge_agg_kernel(const float* __restrict__ vbuf,
                                const int* __restrict__ ei,
                                const float* __restrict__ mrel_l,
                                const float* __restrict__ alpha,
                                const float* __restrict__ s,
                                float* __restrict__ agg)
{
  __shared__ float Mr[DD * DD];
  const int rh = blockIdx.y;
  const int r  = rh >> 2;
  const int hh = rh & 3;
  const float* Mg = mrel_l + (size_t)rh * DD * DD;
  for (int i = threadIdx.x; i < DD * DD; i += blockDim.x) Mr[i] = Mg[i];
  __syncthreads();

  int e = blockIdx.x * blockDim.x + threadIdx.x;
  if (e >= EE) return;
  int sn = ei[(r * 2)     * EE + e];
  int dn = ei[(r * 2 + 1) * EE + e];
  float w = alpha[((size_t)r * EE + e) * HH + hh] / s[dn * HH + hh];

  float vv[DD];
  const float* vr = vbuf + (size_t)sn * HIDC + hh * DD;
  #pragma unroll
  for (int d = 0; d < DD; ++d) vv[d] = vr[d];

  float* ao = agg + (size_t)dn * HIDC + hh * DD;
  for (int f = 0; f < DD; ++f) {
    float vp = 0.0f;
    #pragma unroll
    for (int d = 0; d < DD; ++d) vp += vv[d] * Mr[d * DD + f];
    atomicAdd(&ao[f], w * vp);
  }
}

// skip-combine + residual + LayerNorm, one wave32 per row (8 elems/lane)
__global__ void skip_ln_kernel(float* __restrict__ h,
                               const float* __restrict__ conv_out,
                               const float* __restrict__ skip, int l,
                               const float* __restrict__ gamma,
                               const float* __restrict__ beta)
{
  int row = blockIdx.x * 8 + (threadIdx.x >> 5);
  if (row >= NN) return;
  int lane = threadIdx.x & 31;
  float sk = 1.0f / (1.0f + expf(-skip[l]));

  float y[8];
  float sum = 0.0f;
  #pragma unroll
  for (int j = 0; j < 8; ++j) {
    int c = lane + j * 32;
    float hv = h[(size_t)row * HIDC + c];
    float ov = conv_out[(size_t)row * HIDC + c];
    float yv = hv + (sk * ov + (1.0f - sk) * hv);   // h + hn
    y[j] = yv;
    sum += yv;
  }
  float mu = wave_sum32(sum) * (1.0f / HIDC);
  float vs = 0.0f;
  #pragma unroll
  for (int j = 0; j < 8; ++j) { float d = y[j] - mu; vs += d * d; }
  float var = wave_sum32(vs) * (1.0f / HIDC);
  float rs = rsqrtf(var + 1e-5f);
  #pragma unroll
  for (int j = 0; j < 8; ++j) {
    int c = lane + j * 32;
    h[(size_t)row * HIDC + c] = (y[j] - mu) * rs * gamma[c] + beta[c];
  }
}

// ---------------------------------------------------------------------------
// Small utility kernels
// ---------------------------------------------------------------------------
__global__ void fill_kernel(float* p, float v, int n) {
  int i = blockIdx.x * blockDim.x + threadIdx.x;
  if (i < n) p[i] = v;
}
__global__ void f32_to_bf16_kernel(const float* __restrict__ src,
                                   unsigned short* __restrict__ dst, int n) {
  int i = blockIdx.x * blockDim.x + threadIdx.x;
  if (i < n) dst[i] = f2bf(src[i]);
}
__global__ void pool_count_kernel(const int* __restrict__ batch, float* cnt) {
  int i = blockIdx.x * blockDim.x + threadIdx.x;
  if (i < NN) atomicAdd(&cnt[batch[i]], 1.0f);
}
__global__ void pool_scatter_kernel(const float* __restrict__ h,
                                    const int* __restrict__ batch,
                                    float* gsum, float* gmax) {
  int i = blockIdx.x * blockDim.x + threadIdx.x;
  if (i < NN * HIDC) {
    int node = i / HIDC;
    int c    = i - node * HIDC;
    int b    = batch[node];
    float v  = h[i];
    atomicAdd(&gsum[b * HIDC + c], v);
    atomicMaxF(&gmax[b * HIDC + c], v);
  }
}
__global__ void pool_final_kernel(const float* gsum, const float* gmax,
                                  const float* cnt, float* g) {
  int i = blockIdx.x * blockDim.x + threadIdx.x;
  if (i < BB * 2 * HIDC) {
    int b = i / (2 * HIDC);
    int c = i - b * 2 * HIDC;
    g[i] = (c < HIDC) ? gsum[b * HIDC + c] / fmaxf(cnt[b], 1.0f)
                      : gmax[b * HIDC + (c - HIDC)];
  }
}
// N=1 head: out = A[row,:] . w + bias ; writes logit + sigmoid (wave per row)
__global__ void dot_sig_kernel(const float* __restrict__ A,
                               const float* __restrict__ w,
                               const float* __restrict__ bias,
                               float* outSig, float* outLog, int rows, int K) {
  int row = blockIdx.x * 8 + (threadIdx.x >> 5);
  if (row >= rows) return;
  int lane = threadIdx.x & 31;
  float s = 0.0f;
  for (int k = lane; k < K; k += 32) s += A[(size_t)row * K + k] * w[k];
  s = wave_sum32(s);
  if (lane == 0) {
    float v = s + bias[0];
    outLog[row] = v;
    outSig[row] = 1.0f / (1.0f + expf(-v));
  }
}

// ---------------------------------------------------------------------------
extern "C" void kernel_launch(void* const* d_in, const int* in_sizes, int n_in,
                              void* d_out, int out_size, void* d_ws, size_t ws_size,
                              hipStream_t stream)
{
  (void)in_sizes; (void)n_in; (void)out_size; (void)ws_size;

  const float* x     = (const float*)d_in[0];
  const int*   ei    = (const int*)  d_in[1];
  const int*   batch = (const int*)  d_in[2];
  const float* Wi    = (const float*)d_in[3];
  const float* bi    = (const float*)d_in[4];
  const float* Wk    = (const float*)d_in[5];
  const float* bk    = (const float*)d_in[6];
  const float* Wq    = (const float*)d_in[7];
  const float* bq    = (const float*)d_in[8];
  const float* Wv    = (const float*)d_in[9];
  const float* bv    = (const float*)d_in[10];
  const float* a_rel = (const float*)d_in[11];
  const float* m_rel = (const float*)d_in[12];
  const float* p_rel = (const float*)d_in[13];
  const float* Wout  = (const float*)d_in[14];
  const float* bout  = (const float*)d_in[15];
  const float* skip  = (const float*)d_in[16];
  const float* ln_g  = (const float*)d_in[17];
  const float* ln_b  = (const float*)d_in[18];
  const float* Wg1   = (const float*)d_in[19];
  const float* bg1   = (const float*)d_in[20];
  const float* Wg2   = (const float*)d_in[21];
  const float* bg2   = (const float*)d_in[22];
  const float* Wl1   = (const float*)d_in[23];
  const float* bl1   = (const float*)d_in[24];
  const float* Wl2   = (const float*)d_in[25];
  const float* bl2   = (const float*)d_in[26];
  float* out = (float*)d_out;

  char* wsp = (char*)d_ws;
  auto alloc = [&](size_t bytes) -> void* {
    void* p = (void*)wsp;
    wsp += (bytes + 255) & ~(size_t)255;
    return p;
  };
  float* h    = (float*)alloc((size_t)NN * HIDC * 4);
  float* conv = (float*)alloc((size_t)NN * HIDC * 4);
  float* kb   = (float*)alloc((size_t)NN * HIDC * 4);
  float* qb   = (float*)alloc((size_t)NN * HIDC * 4);
  float* vb   = (float*)alloc((size_t)NN * HIDC * 4);
  float* agg  = (float*)alloc((size_t)NN * HIDC * 4);
  float* hl   = (float*)alloc((size_t)NN * 128 * 4);
  float* alpha= (float*)alloc((size_t)RR * EE * HH * 4);
  float* mbuf = (float*)alloc((size_t)NN * HH * 4);
  float* sbuf = (float*)alloc((size_t)NN * HH * 4);
  float* gsum = (float*)alloc((size_t)BB * HIDC * 4);
  float* gmax = (float*)alloc((size_t)BB * HIDC * 4);
  float* cnt  = (float*)alloc((size_t)BB * 4);
  float* g    = (float*)alloc((size_t)BB * 2 * HIDC * 4);
  float* gg   = (float*)alloc((size_t)BB * HIDC * 4);
  unsigned short* WiB  = (unsigned short*)alloc((size_t)IND * HIDC * 2);
  unsigned short* WkB  = (unsigned short*)alloc((size_t)LL * HIDC * HIDC * 2);
  unsigned short* WqB  = (unsigned short*)alloc((size_t)LL * HIDC * HIDC * 2);
  unsigned short* WvB  = (unsigned short*)alloc((size_t)LL * HIDC * HIDC * 2);
  unsigned short* WoB  = (unsigned short*)alloc((size_t)LL * HIDC * HIDC * 2);
  unsigned short* Wg1B = (unsigned short*)alloc((size_t)2 * HIDC * HIDC * 2);
  unsigned short* Wl1B = (unsigned short*)alloc((size_t)HIDC * 128 * 2);

  auto cvt = [&](const float* s_, unsigned short* d_, int n) {
    f32_to_bf16_kernel<<<(n + 255) / 256, 256, 0, stream>>>(s_, d_, n);
  };
  cvt(Wi,   WiB,  IND * HIDC);
  cvt(Wk,   WkB,  LL * HIDC * HIDC);
  cvt(Wq,   WqB,  LL * HIDC * HIDC);
  cvt(Wv,   WvB,  LL * HIDC * HIDC);
  cvt(Wout, WoB,  LL * HIDC * HIDC);
  cvt(Wg1,  Wg1B, 2 * HIDC * HIDC);
  cvt(Wl1,  Wl1B, HIDC * 128);

  const dim3 gemmGridN(HIDC / 64, (NN + 63) / 64);   // (4, 469)

  // h = relu(x @ Wi + bi)
  gemm_bf16_wmma<0, 1><<<gemmGridN, 128, 0, stream>>>(x, WiB, bi, h, NN, IND, HIDC);

  const dim3 egrid((EE + 127) / 128, RR * HH);

  for (int l = 0; l < LL; ++l) {
    gemm_bf16_wmma<0, 0><<<gemmGridN, 128, 0, stream>>>(
        h, WkB + (size_t)l * HIDC * HIDC, bk + l * HIDC, kb, NN, HIDC, HIDC);
    gemm_bf16_wmma<0, 0><<<gemmGridN, 128, 0, stream>>>(
        h, WqB + (size_t)l * HIDC * HIDC, bq + l * HIDC, qb, NN, HIDC, HIDC);
    gemm_bf16_wmma<0, 0><<<gemmGridN, 128, 0, stream>>>(
        h, WvB + (size_t)l * HIDC * HIDC, bv + l * HIDC, vb, NN, HIDC, HIDC);

    fill_kernel<<<(NN * HH + 255) / 256, 256, 0, stream>>>(mbuf, -INFINITY, NN * HH);
    fill_kernel<<<(NN * HH + 255) / 256, 256, 0, stream>>>(sbuf, 0.0f, NN * HH);
    fill_kernel<<<(NN * HIDC + 255) / 256, 256, 0, stream>>>(agg, 0.0f, NN * HIDC);

    edge_alpha_kernel<<<egrid, 128, 0, stream>>>(
        kb, qb, ei, a_rel + (size_t)l * RR * HH * DD * DD, p_rel + l * RR * HH,
        alpha, mbuf);
    fix_m_kernel<<<(NN * HH + 255) / 256, 256, 0, stream>>>(mbuf, NN * HH);
    edge_exp_kernel<<<(RR * EE * HH + 255) / 256, 256, 0, stream>>>(ei, mbuf, alpha, sbuf);
    edge_agg_kernel<<<egrid, 128, 0, stream>>>(
        vb, ei, m_rel + (size_t)l * RR * HH * DD * DD, alpha, sbuf, agg);

    // conv = gelu(agg) @ Wout + bout   (gelu fused into A-tile conversion)
    gemm_bf16_wmma<1, 0><<<gemmGridN, 128, 0, stream>>>(
        agg, WoB + (size_t)l * HIDC * HIDC, bout + l * HIDC, conv, NN, HIDC, HIDC);

    skip_ln_kernel<<<(NN + 7) / 8, 256, 0, stream>>>(
        h, conv, skip, l, ln_g + l * HIDC, ln_b + l * HIDC);
  }

  // graph pooling
  fill_kernel<<<(BB * HIDC + 255) / 256, 256, 0, stream>>>(gsum, 0.0f, BB * HIDC);
  fill_kernel<<<(BB * HIDC + 255) / 256, 256, 0, stream>>>(gmax, -INFINITY, BB * HIDC);
  fill_kernel<<<1, 256, 0, stream>>>(cnt, 0.0f, BB);
  pool_count_kernel<<<(NN + 255) / 256, 256, 0, stream>>>(batch, cnt);
  pool_scatter_kernel<<<(NN * HIDC + 255) / 256, 256, 0, stream>>>(h, batch, gsum, gmax);
  pool_final_kernel<<<(BB * 2 * HIDC + 255) / 256, 256, 0, stream>>>(gsum, gmax, cnt, g);

  // graph head: vlog
  gemm_bf16_wmma<0, 1><<<dim3(HIDC / 64, (BB + 63) / 64), 128, 0, stream>>>(
      g, Wg1B, bg1, gg, BB, 2 * HIDC, HIDC);
  dot_sig_kernel<<<(BB + 7) / 8, 256, 0, stream>>>(gg, Wg2, bg2, out, out + BB, BB, HIDC);

  // node head: nlog
  gemm_bf16_wmma<0, 1><<<dim3(128 / 64, (NN + 63) / 64), 128, 0, stream>>>(
      h, Wl1B, bl1, hl, NN, HIDC, 128);
  dot_sig_kernel<<<(NN + 7) / 8, 256, 0, stream>>>(
      hl, Wl2, bl2, out + 2 * BB, out + 2 * BB + NN, NN, 128);
}
